// SDN_15496242003980
// MI455X (gfx1250) — matrix operand order repaired and checked by reading, not compiled, not run on hardware
//
#include <hip/hip_runtime.h>
#include <hip/hip_bf16.h>

// ---------------------------------------------------------------------------
// CDNA5 / gfx1250 bf16-WMMA MLP: out = sda(sda(x@W1^T+b1)@W2^T+b2)@W3^T+b3
// wave32, V_WMMA_F32_16X16X32_BF16, f32 accumulate, async-to-LDS staging.
// ---------------------------------------------------------------------------

typedef __attribute__((ext_vector_type(16))) __bf16 bf16x16;
typedef __attribute__((ext_vector_type(8)))  __bf16 bf16x8;
typedef __attribute__((ext_vector_type(8)))  float  f32x8;
typedef __attribute__((ext_vector_type(4)))  float  f32x4;
typedef __attribute__((ext_vector_type(4)))  int    v4i;

union FragAB { bf16x16 v; bf16x8 h[2]; };

// ---------------- CDNA5 async-copy primitives (probe-guarded) --------------
#if defined(__has_builtin)
#  if __has_builtin(__builtin_amdgcn_global_load_async_to_lds_b128)
#    define HAVE_ASYNC_B128 1
#  endif
#  if __has_builtin(__builtin_amdgcn_s_wait_asynccnt)
#    define HAVE_WAIT_ASYNC 1
#  endif
#endif

// 16-byte global -> LDS async copy (ASYNCcnt-tracked).
__device__ __forceinline__ void async_cp16(void* lds_generic, const void* gsrc) {
    // Flat shared address -> 32-bit LDS byte offset (aperture: addr[31:0]).
    unsigned ldsAddr = (unsigned)(size_t)lds_generic;
#if defined(HAVE_ASYNC_B128)
    __builtin_amdgcn_global_load_async_to_lds_b128(
        (__attribute__((address_space(1))) v4i*)(size_t)gsrc,
        (__attribute__((address_space(3))) v4i*)(size_t)ldsAddr,
        0, 0);
#else
    asm volatile("global_load_async_to_lds_b128 %0, %1, off"
                 :: "v"(ldsAddr), "v"(gsrc)
                 : "memory");
#endif
}

template <int N>
__device__ __forceinline__ void wait_asynccnt() {
#if defined(HAVE_WAIT_ASYNC)
    __builtin_amdgcn_s_wait_asynccnt(N);
#else
    asm volatile("s_wait_asynccnt %0" :: "i"(N));
#endif
}

// ---------------------------------------------------------------------------
// f32 -> bf16 bulk convert (8 elems / thread)
// ---------------------------------------------------------------------------
__global__ __launch_bounds__(256)
void cvt_f32_to_bf16(const float* __restrict__ src, __bf16* __restrict__ dst, int n8) {
    int i = blockIdx.x * blockDim.x + threadIdx.x;
    if (i >= n8) return;
    const f32x4* s = reinterpret_cast<const f32x4*>(src) + (size_t)i * 2;
    f32x4 a = s[0];
    f32x4 b = s[1];
    bf16x8 o;
    o[0] = (__bf16)a[0]; o[1] = (__bf16)a[1]; o[2] = (__bf16)a[2]; o[3] = (__bf16)a[3];
    o[4] = (__bf16)b[0]; o[5] = (__bf16)b[1]; o[6] = (__bf16)b[2]; o[7] = (__bf16)b[3];
    reinterpret_cast<bf16x8*>(dst)[i] = o;
}

// ---------------------------------------------------------------------------
// GEMM: C[M,N] = A[M,K] @ W[N,K]^T + bias[N]; A,W bf16 K-major; f32 accum.
//
// Block = 256 threads = 8 wave32 (4m x 2n), 128x128 output tile.
// K-stage = 64: A tile 128x64 (16KB) and B tile 128x64 (16KB) are staged into
// LDS with GLOBAL_LOAD_ASYNC_TO_LDS_B128, double-buffered (64KB LDS total).
// Each wave: 32x64 output = 2x4 accumulators, 16 WMMAs per stage.
//
// LDS layout: row/col r of the stage at byte offset r*128 (64 bf16 of K).
// Fragment reads use this contiguous K order (same for A and B):
//   A 16x32: lane m -> row m, K {0..15};       lane m+16 -> K {16..31}.
//   B 32x16: lane n -> col n, K {0..15};       lane n+16 -> K {16..31}.
// ---------------------------------------------------------------------------
template <bool OUT_BF16>
__global__ __launch_bounds__(256)
void gemm_bf16_wmma(const __bf16* __restrict__ A, const __bf16* __restrict__ W,
                    const float* __restrict__ bias,
                    float* __restrict__ outF, __bf16* __restrict__ outB,
                    int M, int N, int K) {
    __shared__ __align__(16) char smem[64 * 1024];   // A0 A1 B0 B1, 16KB each

    const int tid    = threadIdx.x;
    const int lane   = tid & 31;
    const int wave   = tid >> 5;
    const int wm     = wave & 3;          // 0..3 -> 32-row slab
    const int wn     = wave >> 2;         // 0..1 -> 64-col slab
    const int lh     = lane & 15;
    const bool hiHalf = lane >= 16;

    const int m0 = blockIdx.y * 128 + wm * 32;
    const int n0 = blockIdx.x * 128 + wn * 64;

    f32x8 acc[2][4];
#pragma unroll
    for (int i = 0; i < 2; ++i)
#pragma unroll
        for (int j = 0; j < 4; ++j)
            acc[i][j] = f32x8{0.f, 0.f, 0.f, 0.f, 0.f, 0.f, 0.f, 0.f};

    // Async-copy source pointers: thread t owns 16B chunks {t, t+256, t+512,
    // t+768} of the 16KB stage => row (t>>3)+p*32, chunk-in-row t&7.
    const __bf16* gA = A + (size_t)(blockIdx.y * 128 + (tid >> 3)) * K + (tid & 7) * 8;
    const __bf16* gB = W + (size_t)(blockIdx.x * 128 + (tid >> 3)) * K + (tid & 7) * 8;

    auto issue_stage = [&](int buf, int kk) {
        char* la = smem + buf * 16384 + tid * 16;
        char* lb = smem + 32768 + buf * 16384 + tid * 16;
        const __bf16* ga = gA + kk;
        const __bf16* gb = gB + kk;
#pragma unroll
        for (int p = 0; p < 4; ++p) {
            async_cp16(la + p * 4096, ga + (size_t)p * 32 * K);
            async_cp16(lb + p * 4096, gb + (size_t)p * 32 * K);
        }
    };

    issue_stage(0, 0);

    const int rl0       = wm * 32 + lh;            // local A row, subtile 0
    const int cl0       = wn * 64 + lh;            // local B col, subtile j=0
    const unsigned aSel = hiHalf ? 32u : 0u;       // +16 K elems
    const unsigned bSel = hiHalf ? 32u : 0u;       // +16 K elems

    int cur = 0;
    for (int k = 0; k < K; k += 64, cur ^= 1) {
        if (k + 64 < K) {
            issue_stage(cur ^ 1, k + 64);
            wait_asynccnt<8>();     // this stage's 8 copies done; next 8 in flight
        } else {
            wait_asynccnt<0>();
        }
        __syncthreads();            // whole tile visible to all waves

        const char* sa = smem + cur * 16384;
        const char* sb = smem + 32768 + cur * 16384;
#pragma unroll
        for (int ks = 0; ks < 2; ++ks) {
            FragAB a0, a1, b0, b1, b2, b3;
            const char* ap = sa + rl0 * 128 + ks * 64 + aSel;
            a0.h[0] = *reinterpret_cast<const bf16x8*>(ap);
            a0.h[1] = *reinterpret_cast<const bf16x8*>(ap + 16);
            a1.h[0] = *reinterpret_cast<const bf16x8*>(ap + 16 * 128);
            a1.h[1] = *reinterpret_cast<const bf16x8*>(ap + 16 * 128 + 16);
            const char* bp = sb + cl0 * 128 + ks * 64 + bSel;
            b0.h[0] = *reinterpret_cast<const bf16x8*>(bp);
            b0.h[1] = *reinterpret_cast<const bf16x8*>(bp + 16);
            b1.h[0] = *reinterpret_cast<const bf16x8*>(bp + 16 * 128);
            b1.h[1] = *reinterpret_cast<const bf16x8*>(bp + 16 * 128 + 16);
            b2.h[0] = *reinterpret_cast<const bf16x8*>(bp + 32 * 128);
            b2.h[1] = *reinterpret_cast<const bf16x8*>(bp + 32 * 128 + 16);
            b3.h[0] = *reinterpret_cast<const bf16x8*>(bp + 48 * 128);
            b3.h[1] = *reinterpret_cast<const bf16x8*>(bp + 48 * 128 + 16);

            acc[0][0] = __builtin_amdgcn_wmma_f32_16x16x32_bf16(false, a0.v, false, b0.v, (short)0, acc[0][0], false, false);
            acc[0][1] = __builtin_amdgcn_wmma_f32_16x16x32_bf16(false, a0.v, false, b1.v, (short)0, acc[0][1], false, false);
            acc[0][2] = __builtin_amdgcn_wmma_f32_16x16x32_bf16(false, a0.v, false, b2.v, (short)0, acc[0][2], false, false);
            acc[0][3] = __builtin_amdgcn_wmma_f32_16x16x32_bf16(false, a0.v, false, b3.v, (short)0, acc[0][3], false, false);
            acc[1][0] = __builtin_amdgcn_wmma_f32_16x16x32_bf16(false, a1.v, false, b0.v, (short)0, acc[1][0], false, false);
            acc[1][1] = __builtin_amdgcn_wmma_f32_16x16x32_bf16(false, a1.v, false, b1.v, (short)0, acc[1][1], false, false);
            acc[1][2] = __builtin_amdgcn_wmma_f32_16x16x32_bf16(false, a1.v, false, b2.v, (short)0, acc[1][2], false, false);
            acc[1][3] = __builtin_amdgcn_wmma_f32_16x16x32_bf16(false, a1.v, false, b3.v, (short)0, acc[1][3], false, false);
        }
        __syncthreads();            // done reading buf[cur] before it is refilled
    }

    // Epilogue. C/D layout: VGPR r in lane l -> row = 16*i + (l>=16?8:0) + r,
    // col = n0 + 16*j + (l&15).
    const int rbase = m0 + (hiHalf ? 8 : 0);
#pragma unroll
    for (int i = 0; i < 2; ++i) {
#pragma unroll
        for (int j = 0; j < 4; ++j) {
            const int col = n0 + j * 16 + lh;
            const float bv = bias[col];
#pragma unroll
            for (int r = 0; r < 8; ++r) {
                const int row = rbase + i * 16 + r;
                const float v = acc[i][j][r] + bv;
                if (OUT_BF16) outB[(size_t)row * N + col] = (__bf16)v;
                else          outF[(size_t)row * N + col] = v;
            }
        }
    }
}

// ---------------------------------------------------------------------------
// sda activation, in place on bf16 [rows, H], group=4, alpha=2.
// One 256-thread block per row; H=4096 -> 16 elems (4 groups) per thread.
// ---------------------------------------------------------------------------
__global__ __launch_bounds__(256)
void sda_inplace_bf16(__bf16* __restrict__ hbuf, int H) {
    __shared__ unsigned sPos, sNeg;
    if (threadIdx.x == 0) { sPos = 0xFFFFFFFFu; sNeg = 0xFFFFFFFFu; }
    __syncthreads();

    __bf16* p = hbuf + (size_t)blockIdx.x * H + (size_t)threadIdx.x * 16;
    bf16x8 v0 = reinterpret_cast<bf16x8*>(p)[0];
    bf16x8 v1 = reinterpret_cast<bf16x8*>(p)[1];

    float f[16];
#pragma unroll
    for (int i = 0; i < 8; ++i) { f[i] = (float)v0[i]; f[8 + i] = (float)v1[i]; }

    const unsigned gbase = (unsigned)threadIdx.x * 4u;
#pragma unroll
    for (int g = 0; g < 4; ++g) {
        bool allPos = true, allNeg = true;
#pragma unroll
        for (int e = 0; e < 4; ++e) {
            const float x = f[g * 4 + e];
            allPos = allPos && (x > 0.f);
            allNeg = allNeg && (x < 0.f);
        }
        if (allPos) atomicMin(&sPos, gbase + g);
        if (allNeg) atomicMin(&sNeg, gbase + g);
    }
    __syncthreads();

    const unsigned pi = sPos;
    const unsigned ni = sNeg;
#pragma unroll
    for (int g = 0; g < 4; ++g) {
        const unsigned gi = gbase + g;
        float s = 1.f;
        if (gi == pi) s = 2.f;   // alpha
        if (gi == ni) s = 0.f;
#pragma unroll
        for (int e = 0; e < 4; ++e) f[g * 4 + e] *= s;
    }

#pragma unroll
    for (int i = 0; i < 8; ++i) { v0[i] = (__bf16)f[i]; v1[i] = (__bf16)f[8 + i]; }
    reinterpret_cast<bf16x8*>(p)[0] = v0;
    reinterpret_cast<bf16x8*>(p)[1] = v1;
}

// ---------------------------------------------------------------------------
// Host-side launcher
// ---------------------------------------------------------------------------
extern "C" void kernel_launch(void* const* d_in, const int* in_sizes, int n_in,
                              void* d_out, int out_size, void* d_ws, size_t ws_size,
                              hipStream_t stream) {
    (void)in_sizes; (void)n_in; (void)out_size; (void)ws_size;

    const float* x  = (const float*)d_in[0];   // [8192,1024]
    const float* W1 = (const float*)d_in[1];   // [4096,1024]
    const float* b1 = (const float*)d_in[2];   // [4096]
    const float* W2 = (const float*)d_in[3];   // [4096,4096]
    const float* b2 = (const float*)d_in[4];   // [4096]
    const float* W3 = (const float*)d_in[5];   // [1024,4096]
    const float* b3 = (const float*)d_in[6];   // [1024]
    float* out = (float*)d_out;                // [8192,1024]

    constexpr int Mrows = 8192, Din = 1024, Hdim = 4096, Dout = 1024;

    // Workspace carve-up (192 MB total).
    char* ws = (char*)d_ws;
    __bf16* xb  = (__bf16*)ws; ws += (size_t)Mrows * Din  * 2;   // 16 MB
    __bf16* w1b = (__bf16*)ws; ws += (size_t)Hdim  * Din  * 2;   //  8 MB
    __bf16* w2b = (__bf16*)ws; ws += (size_t)Hdim  * Hdim * 2;   // 32 MB
    __bf16* w3b = (__bf16*)ws; ws += (size_t)Dout  * Hdim * 2;   //  8 MB
    __bf16* h1  = (__bf16*)ws; ws += (size_t)Mrows * Hdim * 2;   // 64 MB
    __bf16* h2  = (__bf16*)ws;                                   // 64 MB

    auto cvt = [&](const float* s, __bf16* d, size_t n) {
        int n8 = (int)(n / 8);
        cvt_f32_to_bf16<<<(n8 + 255) / 256, 256, 0, stream>>>(s, d, n8);
    };
    cvt(x,  xb,  (size_t)Mrows * Din);
    cvt(W1, w1b, (size_t)Hdim  * Din);
    cvt(W2, w2b, (size_t)Hdim  * Hdim);
    cvt(W3, w3b, (size_t)Dout  * Hdim);

    const dim3 blk(256);

    gemm_bf16_wmma<true><<<dim3(Hdim / 128, Mrows / 128), blk, 0, stream>>>(
        xb, w1b, b1, nullptr, h1, Mrows, Hdim, Din);
    sda_inplace_bf16<<<Mrows, 256, 0, stream>>>(h1, Hdim);

    gemm_bf16_wmma<true><<<dim3(Hdim / 128, Mrows / 128), blk, 0, stream>>>(
        h1, w2b, b2, nullptr, h2, Mrows, Hdim, Hdim);
    sda_inplace_bf16<<<Mrows, 256, 0, stream>>>(h2, Hdim);

    gemm_bf16_wmma<false><<<dim3(Dout / 128, Mrows / 128), blk, 0, stream>>>(
        h2, w3b, b3, out, nullptr, Mrows, Dout, Hdim);
}
